// GCN_34961033789882
// MI455X (gfx1250) — matrix operand order repaired and checked by reading, not compiled, not run on hardware
//
#include <hip/hip_runtime.h>

typedef __attribute__((ext_vector_type(16))) _Float16 v16h;
typedef __attribute__((ext_vector_type(8)))  float    v8f;

#define DFEAT 64

// ---------------------------------------------------------------- utilities
__global__ void k_zero(float* __restrict__ p, int n) {
  int i = blockIdx.x * blockDim.x + threadIdx.x;
  if (i < n) p[i] = 0.0f;
}

__global__ void k_count_deg(const int* __restrict__ dst, float* __restrict__ deg, int E) {
  int stride = gridDim.x * blockDim.x;
  for (int e = blockIdx.x * blockDim.x + threadIdx.x; e < E; e += stride)
    unsafeAtomicAdd(&deg[dst[e]], 1.0f);   // hardware global_atomic_add_f32
}

__global__ void k_norm(float* __restrict__ dn, int n) {
  int i = blockIdx.x * blockDim.x + threadIdx.x;
  if (i < n) dn[i] = rsqrtf(dn[i] + 1.0f);
}

// ------------------------------------------------- GEMM: hW = h @ W  (WMMA)
// Also initializes agg = hW * norm^2 (self-loop term of GCNConv).
__global__ void k_gemm_wmma(const float* __restrict__ h,
                            const float* __restrict__ W,
                            const float* __restrict__ norm,
                            float* __restrict__ hW,
                            float* __restrict__ agg,
                            int nrows) {
  __shared__ __align__(32) _Float16 Wt[DFEAT * DFEAT];   // Wt[n*64 + k] = W[k][n]
  int tid = threadIdx.x;
  for (int idx = tid; idx < DFEAT * DFEAT; idx += blockDim.x) {
    int n = idx >> 6, k = idx & 63;
    Wt[idx] = (_Float16)W[k * DFEAT + n];
  }
  __syncthreads();

  const int lane = tid & 31;
  const int wave = tid >> 5;
  const int nn   = lane & 15;     // col within 16-wide tile / row within A tile
  const int hi   = lane >> 4;     // 0 or 1 (lane half)

  // B fragments (wave-invariant): 2 K-chunks x 4 N-tiles
  v16h bf[2][4];
#pragma unroll
  for (int kb2 = 0; kb2 < 2; ++kb2) {
    int kk = kb2 * 32 + hi * 16;  // lanes 0-15: K = kb..kb+15 ; lanes 16-31: K = kb+16..kb+31
#pragma unroll
    for (int nt = 0; nt < 4; ++nt) {
      const _Float16* wp = &Wt[(nt * 16 + nn) * DFEAT + kk];
      v16h b;
#pragma unroll
      for (int j = 0; j < 16; ++j) b[j] = wp[j];
      bf[kb2][nt] = b;
    }
  }

  const int ntiles = (nrows + 15) >> 4;
  for (int t = blockIdx.x * 8 + wave; t < ntiles; t += gridDim.x * 8) {
    const int r0 = t << 4;
    int arow = r0 + nn; if (arow >= nrows) arow = nrows - 1;
    const float* hr = h + (size_t)arow * DFEAT;
    const int koff = hi * 16;     // A layout: lane<16 -> K {0..15}; lane>=16 -> {16..31}

    v8f c0 = {}, c1 = {}, c2 = {}, c3 = {};
#pragma unroll
    for (int kb2 = 0; kb2 < 2; ++kb2) {
      const int kb = kb2 * 32;
      v16h a;
#pragma unroll
      for (int j = 0; j < 8; ++j) {
        a[j]     = (_Float16)hr[kb + koff + j];
        a[8 + j] = (_Float16)hr[kb + koff + 8 + j];
      }
      c0 = __builtin_amdgcn_wmma_f32_16x16x32_f16(false, a, false, bf[kb2][0], (short)0, c0, false, false);
      c1 = __builtin_amdgcn_wmma_f32_16x16x32_f16(false, a, false, bf[kb2][1], (short)0, c1, false, false);
      c2 = __builtin_amdgcn_wmma_f32_16x16x32_f16(false, a, false, bf[kb2][2], (short)0, c2, false, false);
      c3 = __builtin_amdgcn_wmma_f32_16x16x32_f16(false, a, false, bf[kb2][3], (short)0, c3, false, false);
    }

    // C/D layout: lane half selects row-block (0 or 8), VGPR v selects row, lane&15 = col
    const int mbase = hi * 8;
#pragma unroll
    for (int v = 0; v < 8; ++v) {
      const int r = r0 + mbase + v;
      if (r < nrows) {
        const float ns = norm[r];
        const float w2 = ns * ns;
        const size_t base = (size_t)r * DFEAT + nn;
        float vals[4] = {c0[v], c1[v], c2[v], c3[v]};
#pragma unroll
        for (int nt = 0; nt < 4; ++nt) {
          hW[base + nt * 16]  = vals[nt];
          agg[base + nt * 16] = vals[nt] * w2;
        }
      }
    }
  }
}

// -------------------------------------------- edge scatter: agg += msg (atomics)
__global__ void k_scatter(const int* __restrict__ src, const int* __restrict__ dst,
                          const float* __restrict__ norm,
                          const float* __restrict__ hW,
                          float* __restrict__ agg, int E) {
  const int lane = threadIdx.x & 31;
  const int wave = threadIdx.x >> 5;
  const int col  = lane * 2;
  const int wstride = gridDim.x * 8;           // 8 waves per 256-thread block
  for (int e = blockIdx.x * 8 + wave; e < E; e += wstride) {
    const int s = src[e];
    const int d = dst[e];
    const float w = norm[s] * norm[d];
    const float2 v = *(const float2*)(hW + (size_t)s * DFEAT + col);
    float* ap = agg + (size_t)d * DFEAT + col;
    unsafeAtomicAdd(ap,     v.x * w);
    unsafeAtomicAdd(ap + 1, v.y * w);
  }
}

// ------------------------------------------------------------ bias + ReLU
__global__ void k_bias_relu(const float* __restrict__ agg, const float* __restrict__ b,
                            float* __restrict__ out, int total) {
  const int stride = gridDim.x * blockDim.x;
  for (int i = blockIdx.x * blockDim.x + threadIdx.x; i < total; i += stride) {
    float v = agg[i] + b[i & 63];
    out[i] = v > 0.0f ? v : 0.0f;
  }
}

// ------------------------------------------------ per-graph mean + projection
__global__ void k_readout(const float* __restrict__ h, const int* __restrict__ ptr,
                          const float* __restrict__ Wp, const float* __restrict__ bp,
                          float* __restrict__ out) {
  __shared__ float red[DFEAT];
  const int g = blockIdx.x;
  const int d = threadIdx.x;
  const int lo = ptr[g], hi = ptr[g + 1];
  float s = 0.0f;
  for (int i = lo; i < hi; ++i) s += h[(size_t)i * DFEAT + d];
  red[d] = (s / (float)(hi - lo)) * Wp[d];
  __syncthreads();
  for (int off = 32; off >= 1; off >>= 1) {
    if (d < off) red[d] += red[d + off];
    __syncthreads();
  }
  if (d == 0) out[g] = red[0] + bp[0];
}

// --------------------------------------------------------------------------
extern "C" void kernel_launch(void* const* d_in, const int* in_sizes, int n_in,
                              void* d_out, int out_size, void* d_ws, size_t ws_size,
                              hipStream_t stream) {
  const float* x   = (const float*)d_in[0];
  const int*   ei  = (const int*)d_in[1];
  const int*   ptr = (const int*)d_in[2];
  const float* W1  = (const float*)d_in[3];
  const float* b1  = (const float*)d_in[4];
  const float* W2  = (const float*)d_in[5];
  const float* b2  = (const float*)d_in[6];
  const float* W3  = (const float*)d_in[7];
  const float* b3  = (const float*)d_in[8];
  const float* Wp  = (const float*)d_in[9];
  const float* bp  = (const float*)d_in[10];

  const int N = in_sizes[0] / DFEAT;
  const int E = in_sizes[1] / 2;
  const int G = in_sizes[2] - 1;
  const int* src = ei;
  const int* dst = ei + E;

  float* ws   = (float*)d_ws;
  float* norm = ws;                               // N floats (deg, then rsqrt in place)
  float* B0   = norm + N;                         // hW   [N, 64]
  float* B1   = B0 + (size_t)N * DFEAT;           // agg  [N, 64]
  float* B2   = B1 + (size_t)N * DFEAT;           // h    [N, 64]
  float* out  = (float*)d_out;

  // degree + norm
  k_zero<<<(N + 255) / 256, 256, 0, stream>>>(norm, N);
  k_count_deg<<<1024, 256, 0, stream>>>(dst, norm, E);
  k_norm<<<(N + 255) / 256, 256, 0, stream>>>(norm, N);

  const int ntiles      = (N + 15) / 16;
  const int gemm_blocks = (ntiles + 7) / 8;
  const int scat_blocks = 4096;
  const int total       = N * DFEAT;

  const float* hin = x;
  const float* Ws[3] = {W1, W2, W3};
  const float* bs[3] = {b1, b2, b3};
  for (int l = 0; l < 3; ++l) {
    k_gemm_wmma<<<gemm_blocks, 256, 0, stream>>>(hin, Ws[l], norm, B0, B1, N);
    k_scatter<<<scat_blocks, 256, 0, stream>>>(src, dst, norm, B0, B1, E);
    k_bias_relu<<<(total + 255) / 256, 256, 0, stream>>>(B1, bs[l], B2, total);
    hin = B2;
  }
  k_readout<<<G, DFEAT, 0, stream>>>(B2, ptr, Wp, bp, out);
}
